// ClusterNet_48739288875254
// MI455X (gfx1250) — matrix operand order repaired and checked by reading, not compiled, not run on hardware
//
#include <hip/hip_runtime.h>
#include <hip/hip_bf16.h>

typedef __attribute__((ext_vector_type(16))) _Float16 v16h;
typedef __attribute__((ext_vector_type(8)))  float    v8f;

#define BN_EPS 1e-5f

// ---------------- problem sizes ----------------
#define BSZ   512
#define NC1   32
#define NC2   64
#define NC3   256

#define A1_ELEMS (BSZ*6*28*28)      // 2408448
#define X1_ELEMS (BSZ*6*14*14)      // 602112
#define A2_ELEMS (BSZ*16*10*10)     // 819200
#define A3_ELEMS (BSZ*120)          // 61440
#define L3_ELEMS (BSZ*256)          // 131072

#define KPAD  416                   // fc1 K: 400 -> 13*32
#define NPAD  128                   // fc1 N: 120 -> 8*16
#define K3PAD 128                   // logits3 K: 120 -> 4*32

// ---------------- d_out layout (floats, concatenated return order) ----------------
#define OFF_ASSIGN 0
#define OFF_LOSS   (BSZ)                         // 512
#define OFF_ACT4   (OFF_LOSS + 1)                // 513
#define OFF_LOG1   (OFF_ACT4 + BSZ*10)           // 5633
#define OFF_LOG2   (OFF_LOG1 + BSZ*NC1*28*28)    // 12850689

// =======================================================================
// conv1: one block per image, LDS-resident image + weights, + BN stats
// =======================================================================
__global__ __launch_bounds__(256)
void conv1_kernel(const float* __restrict__ inp, const float* __restrict__ w,
                  const float* __restrict__ bias, float* __restrict__ act1,
                  float* __restrict__ gstats) {
    __shared__ float simg[3*32*32];
    __shared__ float sw[6*3*25];
    __shared__ float sb[6];
    __shared__ float ssum[6], ssq[6];
    const int b = blockIdx.x;
    for (int i = threadIdx.x; i < 3*32*32; i += 256) simg[i] = inp[b*3072 + i];
    for (int i = threadIdx.x; i < 450;     i += 256) sw[i]   = w[i];
    if (threadIdx.x < 6) { sb[threadIdx.x] = bias[threadIdx.x];
                           ssum[threadIdx.x] = 0.f; ssq[threadIdx.x] = 0.f; }
    __syncthreads();
    for (int o = threadIdx.x; o < 6*28*28; o += 256) {
        const int oc = o / 784, pos = o % 784;
        const int oh = pos / 28, ow = pos % 28;
        float acc = sb[oc];
        #pragma unroll
        for (int ic = 0; ic < 3; ic++) {
            const float* ip = &simg[ic*1024 + oh*32 + ow];
            const float* wp = &sw[(oc*3 + ic)*25];
            #pragma unroll
            for (int kh = 0; kh < 5; kh++)
                #pragma unroll
                for (int kw = 0; kw < 5; kw++)
                    acc = fmaf(ip[kh*32 + kw], wp[kh*5 + kw], acc);
        }
        act1[b*4704 + o] = acc;
        atomicAdd(&ssum[oc], acc);
        atomicAdd(&ssq[oc],  acc*acc);
    }
    __syncthreads();
    if (threadIdx.x < 6) {
        atomicAdd(&gstats[threadIdx.x],     ssum[threadIdx.x]);
        atomicAdd(&gstats[6 + threadIdx.x], ssq[threadIdx.x]);
    }
}

// =======================================================================
__global__ void bn_finalize_kernel(const float* __restrict__ gstats,
                                   float* __restrict__ bnp, int C, float inv_count) {
    int c = threadIdx.x;
    if (c < C) {
        float m = gstats[c] * inv_count;
        float v = gstats[C + c] * inv_count - m*m;
        bnp[c]     = m;
        bnp[C + c] = rsqrtf(v + BN_EPS);
    }
}

// =======================================================================
// BN1 apply + logits1 (L2 over 6 channels to 32 codewords) + ReLU in place
// =======================================================================
__global__ __launch_bounds__(256)
void bn1_logits1_kernel(float* __restrict__ act1, const float* __restrict__ bnp,
                        const float* __restrict__ g, const float* __restrict__ be,
                        const float* __restrict__ k1s, float* __restrict__ out) {
    __shared__ float sk[NC1*6];
    __shared__ float sm[6], si[6], sg[6], sbb[6];
    for (int i = threadIdx.x; i < NC1*6; i += 256) sk[i] = k1s[i];
    if (threadIdx.x < 6) { sm[threadIdx.x] = bnp[threadIdx.x];
                           si[threadIdx.x] = bnp[6 + threadIdx.x];
                           sg[threadIdx.x] = g[threadIdx.x];
                           sbb[threadIdx.x] = be[threadIdx.x]; }
    __syncthreads();
    const int t = blockIdx.x*256 + threadIdx.x;
    if (t >= BSZ*784) return;
    const int b = t / 784, pos = t % 784;
    float a[6];
    #pragma unroll
    for (int c = 0; c < 6; c++) {
        float raw = act1[b*4704 + c*784 + pos];
        float an  = (raw - sm[c]) * si[c] * sg[c] + sbb[c];
        a[c] = an;
        act1[b*4704 + c*784 + pos] = fmaxf(an, 0.f);
    }
    float* op = out + OFF_LOG1 + (size_t)b*NC1*784 + pos;
    #pragma unroll 4
    for (int k = 0; k < NC1; k++) {
        float d = 0.f;
        #pragma unroll
        for (int c = 0; c < 6; c++) { float df = a[c] - sk[k*6 + c]; d = fmaf(df, df, d); }
        op[(size_t)k*784] = sqrtf(d);
    }
}

// =======================================================================
__global__ __launch_bounds__(256)
void maxpool_kernel(const float* __restrict__ in, float* __restrict__ outp,
                    int BC, int H, int W) {
    const int PH = H >> 1, PW = W >> 1;
    int t = blockIdx.x*256 + threadIdx.x;
    if (t >= BC*PH*PW) return;
    int bc = t / (PH*PW), p = t % (PH*PW);
    int ph = p / PW, pw = p % PW;
    const float* ip = in + (size_t)bc*H*W + (2*ph)*W + 2*pw;
    outp[t] = fmaxf(fmaxf(ip[0], ip[1]), fmaxf(ip[W], ip[W+1]));
}

// =======================================================================
// conv2: one block per image: [6,14,14] -> [16,10,10] + BN stats
// =======================================================================
__global__ __launch_bounds__(256)
void conv2_kernel(const float* __restrict__ x1, const float* __restrict__ w,
                  const float* __restrict__ bias, float* __restrict__ act2,
                  float* __restrict__ gstats) {
    __shared__ float simg[6*14*14];
    __shared__ float sw[16*6*25];
    __shared__ float sb[16];
    __shared__ float ssum[16], ssq[16];
    const int b = blockIdx.x;
    for (int i = threadIdx.x; i < 6*196;   i += 256) simg[i] = x1[b*1176 + i];
    for (int i = threadIdx.x; i < 16*6*25; i += 256) sw[i]   = w[i];
    if (threadIdx.x < 16) { sb[threadIdx.x] = bias[threadIdx.x];
                            ssum[threadIdx.x] = 0.f; ssq[threadIdx.x] = 0.f; }
    __syncthreads();
    for (int o = threadIdx.x; o < 16*100; o += 256) {
        const int oc = o / 100, pos = o % 100;
        const int oh = pos / 10, ow = pos % 10;
        float acc = sb[oc];
        #pragma unroll
        for (int ic = 0; ic < 6; ic++) {
            const float* ip = &simg[ic*196 + oh*14 + ow];
            const float* wp = &sw[(oc*6 + ic)*25];
            #pragma unroll
            for (int kh = 0; kh < 5; kh++)
                #pragma unroll
                for (int kw = 0; kw < 5; kw++)
                    acc = fmaf(ip[kh*14 + kw], wp[kh*5 + kw], acc);
        }
        act2[b*1600 + o] = acc;
        atomicAdd(&ssum[oc], acc);
        atomicAdd(&ssq[oc],  acc*acc);
    }
    __syncthreads();
    if (threadIdx.x < 16) {
        atomicAdd(&gstats[threadIdx.x],      ssum[threadIdx.x]);
        atomicAdd(&gstats[16 + threadIdx.x], ssq[threadIdx.x]);
    }
}

// =======================================================================
// BN2 apply + logits2 (64 codewords x 16 channels) + ReLU in place
// =======================================================================
__global__ __launch_bounds__(256)
void bn2_logits2_kernel(float* __restrict__ act2, const float* __restrict__ bnp,
                        const float* __restrict__ g, const float* __restrict__ be,
                        const float* __restrict__ k2s, float* __restrict__ out) {
    __shared__ float sk[NC2*16];
    __shared__ float sm[16], si[16], sg[16], sbb[16];
    for (int i = threadIdx.x; i < NC2*16; i += 256) sk[i] = k2s[i];
    if (threadIdx.x < 16) { sm[threadIdx.x] = bnp[threadIdx.x];
                            si[threadIdx.x] = bnp[16 + threadIdx.x];
                            sg[threadIdx.x] = g[threadIdx.x];
                            sbb[threadIdx.x] = be[threadIdx.x]; }
    __syncthreads();
    const int t = blockIdx.x*256 + threadIdx.x;
    if (t >= BSZ*100) return;
    const int b = t / 100, pos = t % 100;
    float a[16];
    #pragma unroll
    for (int c = 0; c < 16; c++) {
        float raw = act2[b*1600 + c*100 + pos];
        float an  = (raw - sm[c]) * si[c] * sg[c] + sbb[c];
        a[c] = an;
        act2[b*1600 + c*100 + pos] = fmaxf(an, 0.f);
    }
    float* op = out + OFF_LOG2 + (size_t)b*NC2*100 + pos;
    for (int k = 0; k < NC2; k++) {
        float d = 0.f;
        #pragma unroll
        for (int c = 0; c < 16; c++) { float df = a[c] - sk[k*16 + c]; d = fmaf(df, df, d); }
        op[(size_t)k*100] = sqrtf(d);
    }
}

// =======================================================================
// pool2 + pack to f16 A-matrix [512, 416] (pad pre-zeroed)
// =======================================================================
__global__ __launch_bounds__(256)
void pool2_pack_kernel(const float* __restrict__ act2, _Float16* __restrict__ Ah) {
    int t = blockIdx.x*256 + threadIdx.x;
    if (t >= BSZ*400) return;
    int b = t / 400, r = t % 400;
    int c = r / 25, p = r % 25, ph = p / 5, pw = p % 5;
    const float* ip = act2 + (size_t)b*1600 + c*100 + (2*ph)*10 + 2*pw;
    float m = fmaxf(fmaxf(ip[0], ip[1]), fmaxf(ip[10], ip[11]));
    Ah[(size_t)b*KPAD + r] = (_Float16)m;
}

// =======================================================================
// fc1_w [120,400] -> fragment-swizzled f16 B: [nt][kc][lane][16 halves]
// element: k = kc*32 + (j&8)*2 + (lane>>4)*8 + (j&7) ; n = nt*16 + (lane&15)
// so each lane's WMMA B-fragment is one contiguous 32-byte load.
// =======================================================================
__global__ __launch_bounds__(256)
void packB_fc1_kernel(const float* __restrict__ fc1_w, _Float16* __restrict__ Bswz) {
    int t = blockIdx.x*256 + threadIdx.x;
    if (t >= (NPAD/16)*13*32*16) return;
    int nt  = t / (13*512);
    int r   = t % (13*512);
    int kc  = r / 512;
    int r2  = r % 512;
    int lane = r2 / 16, j = r2 % 16;
    int k = kc*32 + (j & 8)*2 + (lane >> 4)*8 + (j & 7);
    int n = nt*16 + (lane & 15);
    Bswz[t] = (k < 400 && n < 120) ? (_Float16)fc1_w[n*400 + k] : (_Float16)0.f;
}

// =======================================================================
// fc1 GEMM via WMMA: one wave32 per 16x16 tile; 13 k-steps.
// =======================================================================
__global__ __launch_bounds__(32)
void fc1_wmma_kernel(const _Float16* __restrict__ Ah, const _Float16* __restrict__ Bswz,
                     const float* __restrict__ fc1_b, float* __restrict__ act3) {
    const int nt = blockIdx.x;        // 0..7
    const int mt = blockIdx.y;        // 0..31
    const int lane = threadIdx.x;
    const int h16  = lane >> 4;
    const int l    = lane & 15;

    v8f c = {};
    const _Float16* arow = Ah + (size_t)(mt*16 + l)*KPAD + h16*8;
    const _Float16* brow = Bswz + ((size_t)(nt*13)*32 + lane)*16;

    for (int kc = 0; kc < 13; kc++) {
        v16h a, bm;
        const _Float16* ap = arow + kc*32;
        __builtin_prefetch(ap + 32, 0, 1);          // next k-step A row
        #pragma unroll
        for (int j = 0; j < 8; j++) { a[j] = ap[j]; a[8 + j] = ap[16 + j]; }
        bm = *(const v16h*)(brow + (size_t)kc*32*16);
        c = __builtin_amdgcn_wmma_f32_16x16x32_f16(false, a, false, bm,
                                                   (short)0, c, false, false);
    }
    const int col = nt*16 + l;
    if (col < 120) {
        const float bias = fc1_b[col];
        const int row0 = mt*16 + h16*8;
        #pragma unroll
        for (int r = 0; r < 8; r++)
            act3[(size_t)(row0 + r)*120 + col] = c[r] + bias;
    }
}

// =======================================================================
// pack act3 -> f16 [512, 128] (zero pad), for logits3 GEMM
// =======================================================================
__global__ __launch_bounds__(256)
void pack_act3_kernel(const float* __restrict__ act3, _Float16* __restrict__ Ah3) {
    int t = blockIdx.x*256 + threadIdx.x;
    if (t >= BSZ*K3PAD) return;
    int b = t / K3PAD, j = t % K3PAD;
    Ah3[t] = (j < 120) ? (_Float16)act3[(size_t)b*120 + j] : (_Float16)0.f;
}

// row norms of the f16-rounded act3 (consistent with the WMMA dot)
__global__ __launch_bounds__(256)
void norm_a_kernel(const _Float16* __restrict__ Ah3, float* __restrict__ na) {
    int b = blockIdx.x*256 + threadIdx.x;
    if (b >= BSZ) return;
    const _Float16* p = Ah3 + (size_t)b*K3PAD;
    float s = 0.f;
    #pragma unroll 4
    for (int j = 0; j < 120; j++) { float v = (float)p[j]; s = fmaf(v, v, s); }
    na[b] = s;
}

// =======================================================================
// k3s [256,120] -> fragment-swizzled f16 B [nt][kc][lane][16], + f16 norms
// =======================================================================
__global__ __launch_bounds__(256)
void packB_k3_kernel(const float* __restrict__ k3s, _Float16* __restrict__ Kswz) {
    int t = blockIdx.x*256 + threadIdx.x;
    if (t >= (NC3/16)*4*32*16) return;
    int nt  = t / (4*512);
    int r   = t % (4*512);
    int kc  = r / 512;
    int r2  = r % 512;
    int lane = r2 / 16, j = r2 % 16;
    int k = kc*32 + (j & 8)*2 + (lane >> 4)*8 + (j & 7);
    int n = nt*16 + (lane & 15);
    Kswz[t] = (k < 120) ? (_Float16)k3s[n*120 + k] : (_Float16)0.f;
}

__global__ __launch_bounds__(256)
void norm_k_kernel(const float* __restrict__ k3s, float* __restrict__ nk) {
    int n = blockIdx.x*256 + threadIdx.x;
    if (n >= NC3) return;
    const float* p = k3s + (size_t)n*120;
    float s = 0.f;
    #pragma unroll 4
    for (int j = 0; j < 120; j++) { float v = (float)(_Float16)p[j]; s = fmaf(v, v, s); }
    nk[n] = s;
}

// =======================================================================
// logits3 via WMMA: dot = a.k (f16 in, f32 acc); ||a-k|| = sqrt(na+nk-2dot)
// M=512 (32 tiles) x N=256 (16 tiles), 4 k-steps.
// =======================================================================
__global__ __launch_bounds__(32)
void logits3_wmma_kernel(const _Float16* __restrict__ Ah3, const _Float16* __restrict__ Kswz,
                         const float* __restrict__ na, const float* __restrict__ nk,
                         float* __restrict__ lg3) {
    const int nt = blockIdx.x;        // 0..15
    const int mt = blockIdx.y;        // 0..31
    const int lane = threadIdx.x;
    const int h16  = lane >> 4;
    const int l    = lane & 15;

    v8f c = {};
    const _Float16* arow = Ah3 + (size_t)(mt*16 + l)*K3PAD + h16*8;
    const _Float16* brow = Kswz + ((size_t)(nt*4)*32 + lane)*16;

    #pragma unroll
    for (int kc = 0; kc < 4; kc++) {
        v16h a, bm;
        const _Float16* ap = arow + kc*32;
        #pragma unroll
        for (int j = 0; j < 8; j++) { a[j] = ap[j]; a[8 + j] = ap[16 + j]; }
        bm = *(const v16h*)(brow + (size_t)kc*32*16);
        c = __builtin_amdgcn_wmma_f32_16x16x32_f16(false, a, false, bm,
                                                   (short)0, c, false, false);
    }
    const int col = nt*16 + l;
    const float nkc = nk[col];
    const int row0 = mt*16 + h16*8;
    #pragma unroll
    for (int r = 0; r < 8; r++) {
        float d2 = na[row0 + r] + nkc - 2.f*c[r];
        lg3[(size_t)(row0 + r)*NC3 + col] = sqrtf(fmaxf(d2, 0.f));
    }
}

// =======================================================================
// act4 = relu(act3) @ fc2_w.T + fc2_b   -> d_out[OFF_ACT4]
// =======================================================================
__global__ __launch_bounds__(256)
void fc2_kernel(const float* __restrict__ act3, const float* __restrict__ w,
                const float* __restrict__ bias, float* __restrict__ out) {
    int t = blockIdx.x*256 + threadIdx.x;
    if (t >= BSZ*10) return;
    int b = t / 10, n = t % 10;
    float acc = bias[n];
    const float* ap = act3 + (size_t)b*120;
    const float* wp = w + (size_t)n*120;
    #pragma unroll 4
    for (int j = 0; j < 120; j++) acc = fmaf(fmaxf(ap[j], 0.f), wp[j], acc);
    out[OFF_ACT4 + t] = acc;
}

// =======================================================================
// greedy_assign: one 512-thread workgroup, thread i owns row i.
// Cached per-row best; only rows whose best column was the winner rescan.
// =======================================================================
__global__ __launch_bounds__(512)
void greedy_kernel(const float* __restrict__ lg3, float* __restrict__ out) {
    __shared__ float logc[NC3];
    __shared__ int   counts[NC3];
    __shared__ float sval[BSZ];
    __shared__ int   sidx[BSZ];
    __shared__ int   s_wi, s_wk;

    const int i = threadIdx.x;
    if (i < NC3) { logc[i] = 0.f; counts[i] = 0; }

    const float* row = lg3 + (size_t)i*NC3;
    float bestv = -__builtin_inff();
    int   bestk = 0;
    for (int k = 0; k < NC3; k++) {
        float v = 0.1f * row[k];
        if (v > bestv) { bestv = v; bestk = k; }
    }
    bool assigned = false;
    int  myAssign = 0;
    __syncthreads();

    for (int step = 0; step < BSZ; step++) {
        sval[i] = assigned ? -__builtin_inff() : bestv;
        sidx[i] = i;
        __syncthreads();
        #pragma unroll
        for (int off = 256; off > 0; off >>= 1) {
            if (i < off) {
                float v2 = sval[i + off];
                int   i2 = sidx[i + off];
                if (v2 > sval[i] || (v2 == sval[i] && i2 < sidx[i])) {
                    sval[i] = v2; sidx[i] = i2;
                }
            }
            __syncthreads();
        }
        if (i == 0) s_wi = sidx[0];
        __syncthreads();
        const int wi = s_wi;
        if (i == wi) s_wk = bestk;
        __syncthreads();
        const int wk = s_wk;
        if (i == wi) { assigned = true; myAssign = wk; }
        if (i == 0)  { counts[wk] += 1; logc[wk] = __logf((float)counts[wk] + 1.f); }
        __syncthreads();
        if (!assigned && bestk == wk) {
            bestv = -__builtin_inff(); bestk = 0;
            for (int k = 0; k < NC3; k++) {
                float v = 0.1f * row[k] - logc[k];
                if (v > bestv) { bestv = v; bestk = k; }
            }
        }
        __syncthreads();
    }

    out[OFF_ASSIGN + i] = (float)myAssign;
    sval[i] = -0.1f * row[myAssign];
    __syncthreads();
    #pragma unroll
    for (int off = 256; off > 0; off >>= 1) {
        if (i < off) sval[i] += sval[i + off];
        __syncthreads();
    }
    if (i == 0) out[OFF_LOSS] = sval[0] / (float)BSZ;
}

// =======================================================================
extern "C" void kernel_launch(void* const* d_in, const int* in_sizes, int n_in,
                              void* d_out, int out_size, void* d_ws, size_t ws_size,
                              hipStream_t stream) {
    const float* inp     = (const float*)d_in[0];
    const float* conv1_w = (const float*)d_in[1];
    const float* conv1_b = (const float*)d_in[2];
    const float* bn1_g   = (const float*)d_in[3];
    const float* bn1_b   = (const float*)d_in[4];
    const float* conv2_w = (const float*)d_in[5];
    const float* conv2_b = (const float*)d_in[6];
    const float* bn2_g   = (const float*)d_in[7];
    const float* bn2_b   = (const float*)d_in[8];
    const float* fc1_w   = (const float*)d_in[9];
    const float* fc1_b   = (const float*)d_in[10];
    const float* fc2_w   = (const float*)d_in[11];
    const float* fc2_b   = (const float*)d_in[12];
    const float* k1s     = (const float*)d_in[13];
    const float* k2s     = (const float*)d_in[14];
    const float* k3s     = (const float*)d_in[15];
    float* out = (float*)d_out;

    // ---- workspace carve-up ----
    float* wsf     = (float*)d_ws;
    float* f_act1  = wsf;                         // A1_ELEMS
    float* f_x1    = f_act1 + A1_ELEMS;           // X1_ELEMS
    float* f_act2  = f_x1   + X1_ELEMS;           // A2_ELEMS
    float* f_act3  = f_act2 + A2_ELEMS;           // A3_ELEMS
    float* f_lg3   = f_act3 + A3_ELEMS;           // L3_ELEMS
    float* f_st1   = f_lg3  + L3_ELEMS;           // 12
    float* f_bn1   = f_st1  + 12;                 // 12
    float* f_st2   = f_bn1  + 12;                 // 32
    float* f_bn2   = f_st2  + 32;                 // 32
    float* f_na    = f_bn2  + 32;                 // 512
    float* f_nk    = f_na   + BSZ;                // 256
    _Float16* Ah   = (_Float16*)(f_nk + NC3);     // 512*416
    _Float16* Bswz = Ah   + (size_t)BSZ*KPAD;     // 8*13*32*16 = 53248
    _Float16* Ah3  = Bswz + (size_t)53248;        // 512*128   = 65536
    _Float16* Kswz = Ah3  + (size_t)BSZ*K3PAD;    // 16*4*32*16 = 32768

    hipMemsetAsync(f_st1, 0, (12 + 12 + 32 + 32) * sizeof(float), stream);
    hipMemsetAsync(Ah, 0, (size_t)BSZ*KPAD*sizeof(_Float16), stream);

    // stage 1: conv1 + BN1
    conv1_kernel<<<BSZ, 256, 0, stream>>>(inp, conv1_w, conv1_b, f_act1, f_st1);
    bn_finalize_kernel<<<1, 32, 0, stream>>>(f_st1, f_bn1, 6, 1.f/(BSZ*784.f));
    bn1_logits1_kernel<<<(BSZ*784 + 255)/256, 256, 0, stream>>>(
        f_act1, f_bn1, bn1_g, bn1_b, k1s, out);
    maxpool_kernel<<<(BSZ*6*196 + 255)/256, 256, 0, stream>>>(f_act1, f_x1, BSZ*6, 28, 28);

    // stage 2: conv2 + BN2
    conv2_kernel<<<BSZ, 256, 0, stream>>>(f_x1, conv2_w, conv2_b, f_act2, f_st2);
    bn_finalize_kernel<<<1, 32, 0, stream>>>(f_st2, f_bn2, 16, 1.f/(BSZ*100.f));
    bn2_logits2_kernel<<<(BSZ*100 + 255)/256, 256, 0, stream>>>(
        f_act2, f_bn2, bn2_g, bn2_b, k2s, out);
    pool2_pack_kernel<<<(BSZ*400 + 255)/256, 256, 0, stream>>>(f_act2, Ah);

    // stage 3: fc1 via WMMA (fragment-swizzled B)
    packB_fc1_kernel<<<(53248 + 255)/256, 256, 0, stream>>>(fc1_w, Bswz);
    {
        dim3 grid(NPAD/16, BSZ/16);
        fc1_wmma_kernel<<<grid, 32, 0, stream>>>(Ah, Bswz, fc1_b, f_act3);
    }

    // stage 4: heads — act4 (VALU, N=10) and logits3 (WMMA GEMM + norm epilogue)
    fc2_kernel<<<(BSZ*10 + 255)/256, 256, 0, stream>>>(f_act3, fc2_w, fc2_b, out);
    pack_act3_kernel<<<(BSZ*K3PAD + 255)/256, 256, 0, stream>>>(f_act3, Ah3);
    norm_a_kernel<<<(BSZ + 255)/256, 256, 0, stream>>>(Ah3, f_na);
    packB_k3_kernel<<<(32768 + 255)/256, 256, 0, stream>>>(k3s, Kswz);
    norm_k_kernel<<<(NC3 + 255)/256, 256, 0, stream>>>(k3s, f_nk);
    {
        dim3 grid(NC3/16, BSZ/16);
        logits3_wmma_kernel<<<grid, 32, 0, stream>>>(Ah3, Kswz, f_na, f_nk, f_lg3);
    }

    // stage 5: sequential greedy assignment (single workgroup)
    greedy_kernel<<<1, BSZ, 0, stream>>>(f_lg3, out);
}
